// GCN_50723563766325
// MI455X (gfx1250) — matrix operand order repaired and checked by reading, not compiled, not run on hardware
//
#include <hip/hip_runtime.h>

typedef __bf16 v16bf __attribute__((ext_vector_type(16)));
typedef float  v8f   __attribute__((ext_vector_type(8)));

#define HID 256
#define FIN 400
#define KP1 416            // FIN padded to multiple of 32
#define NPG 400            // nodes per graph

__device__ __forceinline__ unsigned short f2bf(float f) {
    unsigned int u = __float_as_uint(f);
    u += 0x7fffu + ((u >> 16) & 1u);     // round-to-nearest-even
    return (unsigned short)(u >> 16);
}
__device__ __forceinline__ float leaky(float v) { return v >= 0.f ? v : 0.01f * v; }

// ---------------- utility fills / norm ----------------
__global__ void k_fill(float* p, float v, long n) {
    long i = (long)blockIdx.x * blockDim.x + threadIdx.x;
    long s = (long)gridDim.x * blockDim.x;
    for (; i < n; i += s) p[i] = v;
}

__global__ void k_deg(const int* __restrict__ dst, float* __restrict__ deg, long E) {
    long i = (long)blockIdx.x * blockDim.x + threadIdx.x;
    if (i < E) atomicAdd(&deg[dst[i]], 1.0f);
}

__global__ void k_rsq(float* d, long n) {
    long i = (long)blockIdx.x * blockDim.x + threadIdx.x;
    if (i < n) { float v = d[i]; d[i] = v > 0.f ? rsqrtf(v) : 0.f; }
}

// ---------------- fp32 -> bf16 conversions ----------------
// x [M][Kin] row-major fp32 -> xb [M][Kp] bf16, zero-padded K
__global__ void k_cvt_pad(const float* __restrict__ x, unsigned short* __restrict__ xb,
                          long M, int Kin, int Kp) {
    long i = (long)blockIdx.x * blockDim.x + threadIdx.x;
    long n = M * (long)Kp;
    long s = (long)gridDim.x * blockDim.x;
    for (; i < n; i += s) {
        int  k = (int)(i % Kp);
        long r = i / Kp;
        xb[i] = (k < Kin) ? f2bf(x[r * (long)Kin + k]) : (unsigned short)0;
    }
}

// W [K][N] fp32 -> Wt [N][Kp] bf16 (transposed, K padded)
__global__ void k_cvt_wt(const float* __restrict__ W, unsigned short* __restrict__ Wt,
                         int K, int N, int Kp) {
    int i = blockIdx.x * blockDim.x + threadIdx.x;
    int tot = N * Kp;
    if (i < tot) {
        int k = i % Kp, n = i / Kp;
        Wt[i] = (k < K) ? f2bf(W[(size_t)k * N + n]) : (unsigned short)0;
    }
}

// ---------------- bf16 WMMA GEMM: C[M][N] = A[M][K] @ Bt[N][K]^T ----------------
// One 16x64 output tile per wave (4 accumulators, A-fragment reused 4x per K-step).
// 8 waves (256 threads) per block. N must be a multiple of 64, K of 32, M of 16.
__global__ void __launch_bounds__(256)
k_gemm(const unsigned short* __restrict__ A, const unsigned short* __restrict__ Bt,
       float* __restrict__ C, int M, int N, int K) {
    int lane = threadIdx.x & 31;
    int wave = threadIdx.x >> 5;
    int w    = blockIdx.x * 8 + wave;
    int ngrp = N >> 6;                         // 64-wide column groups
    int mt   = w / ngrp, ng = w % ngrp;
    if (mt >= (M >> 4)) return;                // wave-uniform guard
    int m0 = mt << 4, n0 = ng << 6;
    int lh = lane >> 4;                        // lane half: 0 or 1
    int ll = lane & 15;

    // A: lanes 0-15 hold K 0..15, lanes 16-31 hold K 16..31
    const unsigned short* arow = A + (size_t)(m0 + ll) * K + lh * 16;
    // B: lanes 0-15 hold K 0..15, lanes 16-31 hold K 16..31 (column n contiguous in Bt)
    const unsigned short* bcol = Bt + (size_t)(n0 + ll) * K + lh * 16;
    const size_t bstep = (size_t)16 * K;       // next 16-column group of Bt

    union U { v16bf v; uint4 u[2]; };
    v8f c0 = {}, c1 = {}, c2 = {}, c3 = {};
    for (int k0 = 0; k0 < K; k0 += 32) {
        U a, b0, b1, b2, b3;
        a.u[0]  = *reinterpret_cast<const uint4*>(arow + k0);        // 8 bf16
        a.u[1]  = *reinterpret_cast<const uint4*>(arow + k0 + 8);    // 8 bf16
        b0.u[0] = *reinterpret_cast<const uint4*>(bcol + k0);
        b0.u[1] = *reinterpret_cast<const uint4*>(bcol + k0 + 8);
        b1.u[0] = *reinterpret_cast<const uint4*>(bcol + bstep + k0);
        b1.u[1] = *reinterpret_cast<const uint4*>(bcol + bstep + k0 + 8);
        b2.u[0] = *reinterpret_cast<const uint4*>(bcol + 2 * bstep + k0);
        b2.u[1] = *reinterpret_cast<const uint4*>(bcol + 2 * bstep + k0 + 8);
        b3.u[0] = *reinterpret_cast<const uint4*>(bcol + 3 * bstep + k0);
        b3.u[1] = *reinterpret_cast<const uint4*>(bcol + 3 * bstep + k0 + 8);
        c0 = __builtin_amdgcn_wmma_f32_16x16x32_bf16(false, a.v, false, b0.v, (short)0, c0, false, false);
        c1 = __builtin_amdgcn_wmma_f32_16x16x32_bf16(false, a.v, false, b1.v, (short)0, c1, false, false);
        c2 = __builtin_amdgcn_wmma_f32_16x16x32_bf16(false, a.v, false, b2.v, (short)0, c2, false, false);
        c3 = __builtin_amdgcn_wmma_f32_16x16x32_bf16(false, a.v, false, b3.v, (short)0, c3, false, false);
    }
    // C/D: VGPR r -> row m0+r (lanes 0-15) / m0+r+8 (lanes 16-31), col group + (lane&15)
    float* crow = C + (size_t)(m0 + lh * 8) * N + n0 + ll;
#pragma unroll
    for (int r = 0; r < 8; ++r) {
        crow[(size_t)r * N]      = c0[r];
        crow[(size_t)r * N + 16] = c1[r];
        crow[(size_t)r * N + 32] = c2[r];
        crow[(size_t)r * N + 48] = c3[r];
    }
}

// ---------------- edge scatter: agg[dst] += h[src] * dinv[src]*dinv[dst] ----------------
__global__ void k_scatter(const int* __restrict__ src, const int* __restrict__ dst,
                          const float* __restrict__ dinv, const float* __restrict__ h,
                          float* __restrict__ agg, long E) {
    long t = (long)blockIdx.x * blockDim.x + threadIdx.x;
    long total = E * (HID / 4);
    if (t >= total) return;
    long e  = t >> 6;                 // HID/4 == 64 chunks per edge
    int  c4 = (int)(t & 63) << 2;
    int  s = src[e], d = dst[e];
    float nrm = dinv[s] * dinv[d];
    float4 hv = *reinterpret_cast<const float4*>(h + (size_t)s * HID + c4);
    float* out = agg + (size_t)d * HID + c4;
    atomicAdd(out + 0, hv.x * nrm);
    atomicAdd(out + 1, hv.y * nrm);
    atomicAdd(out + 2, hv.z * nrm);
    atomicAdd(out + 3, hv.w * nrm);
}

// ---------------- finalize: + self-loop + bias (+leaky), emit fp32 and/or bf16 ----------------
__global__ void k_finalize(const float* __restrict__ agg, const float* __restrict__ h,
                           const float* __restrict__ dinv, const float* __restrict__ bias,
                           float* outf, unsigned short* outb, long Nn, int do_leaky) {
    long t = (long)blockIdx.x * blockDim.x + threadIdx.x;
    long total = Nn * HID;
    if (t >= total) return;
    long i = t >> 8;
    int  c = (int)(t & 255);
    float di = dinv[i];
    float v = agg[t] + h[t] * di * di + bias[c];
    if (do_leaky) v = leaky(v);
    if (outf) outf[t] = v;
    if (outb) outb[t] = f2bf(v);
}

// ---------------- mean pool over 400 nodes per graph ----------------
__global__ void k_pool(const float* __restrict__ h, float* __restrict__ g) {
    int gi = blockIdx.x, c = threadIdx.x;          // 256 threads = HID
    const float* base = h + (size_t)gi * NPG * HID + c;
    float s = 0.f;
    for (int i = 0; i < NPG; ++i) s += base[(size_t)i * HID];
    g[(size_t)gi * HID + c] = s * (1.0f / NPG);
}

// ---------------- tiny MLP head ----------------
__global__ void k_mlp(const float* __restrict__ in, const float* __restrict__ W,
                      const float* __restrict__ bias, float* __restrict__ out,
                      int B, int K, int N, int do_leaky) {
    int t = blockIdx.x * blockDim.x + threadIdx.x;
    if (t >= B * N) return;
    int b = t / N, n = t % N;
    float s = bias[n];
    for (int k = 0; k < K; ++k) s += in[(size_t)b * K + k] * W[(size_t)k * N + n];
    out[t] = do_leaky ? leaky(s) : s;
}

extern "C" void kernel_launch(void* const* d_in, const int* in_sizes, int n_in,
                              void* d_out, int out_size, void* d_ws, size_t ws_size,
                              hipStream_t stream) {
    const float* x    = (const float*)d_in[0];
    const int*   ei   = (const int*)d_in[1];
    const float* W0   = (const float*)d_in[2];
    const float* b0   = (const float*)d_in[3];
    const float* W1   = (const float*)d_in[4];
    const float* b1   = (const float*)d_in[5];
    const float* W2   = (const float*)d_in[6];
    const float* b2   = (const float*)d_in[7];
    const float* fcW0 = (const float*)d_in[8];
    const float* fcb0 = (const float*)d_in[9];
    const float* fcW1 = (const float*)d_in[10];
    const float* fcb1 = (const float*)d_in[11];
    const float* fcW2 = (const float*)d_in[12];
    const float* fcb2 = (const float*)d_in[13];

    const long Nn = (long)in_sizes[0] / FIN;      // 51200
    const long E  = (long)in_sizes[1] / 2;        // 1638400
    const int  B  = (int)(Nn / NPG);              // 128
    const int* src = ei;
    const int* dst = ei + E;

    char* ws = (char*)d_ws;
    size_t off = 0;
    auto alloc = [&](size_t bytes) -> void* {
        void* p = ws + off;
        off = (off + bytes + 255) & ~(size_t)255;
        return p;
    };
    unsigned short* xb   = (unsigned short*)alloc((size_t)Nn * KP1 * 2);
    unsigned short* wt0  = (unsigned short*)alloc((size_t)HID * KP1 * 2);
    unsigned short* wt1  = (unsigned short*)alloc((size_t)HID * HID * 2);
    unsigned short* wt2  = (unsigned short*)alloc((size_t)HID * HID * 2);
    float*          dinv = (float*)alloc((size_t)Nn * 4);
    float*          hlin = (float*)alloc((size_t)Nn * HID * 4);
    float*          agg  = (float*)alloc((size_t)Nn * HID * 4);
    unsigned short* hbf  = (unsigned short*)alloc((size_t)Nn * HID * 2);
    float*          gbuf = (float*)alloc((size_t)B * HID * 4);
    float*          t0   = (float*)alloc((size_t)B * 128 * 4);
    float*          t1   = (float*)alloc((size_t)B * 64 * 4);

    const int TB = 256;
    long nh = Nn * HID;

    // degree (self-loop weight 1 -> init 1.0) and dinv
    k_fill<<<2048, TB, 0, stream>>>(dinv, 1.0f, Nn);
    k_deg<<<(unsigned)((E + TB - 1) / TB), TB, 0, stream>>>(dst, dinv, E);
    k_rsq<<<(unsigned)((Nn + TB - 1) / TB), TB, 0, stream>>>(dinv, Nn);

    // bf16 conversions
    k_cvt_pad<<<4096, TB, 0, stream>>>(x, xb, Nn, FIN, KP1);
    k_cvt_wt<<<(HID * KP1 + TB - 1) / TB, TB, 0, stream>>>(W0, wt0, FIN, HID, KP1);
    k_cvt_wt<<<(HID * HID + TB - 1) / TB, TB, 0, stream>>>(W1, wt1, HID, HID, HID);
    k_cvt_wt<<<(HID * HID + TB - 1) / TB, TB, 0, stream>>>(W2, wt2, HID, HID, HID);

    unsigned gemmGrid = (unsigned)(((Nn / 16) * (HID / 64)) / 8);   // 16x64 tile per wave
    unsigned scatGrid = (unsigned)((E * (HID / 4) + TB - 1) / TB);
    unsigned finGrid  = (unsigned)((nh + TB - 1) / TB);

    // ---- layer 0 ----
    k_gemm<<<gemmGrid, TB, 0, stream>>>(xb, wt0, hlin, (int)Nn, HID, KP1);
    k_fill<<<4096, TB, 0, stream>>>(agg, 0.0f, nh);
    k_scatter<<<scatGrid, TB, 0, stream>>>(src, dst, dinv, hlin, agg, E);
    k_finalize<<<finGrid, TB, 0, stream>>>(agg, hlin, dinv, b0, nullptr, hbf, Nn, 1);

    // ---- layer 1 ----
    k_gemm<<<gemmGrid, TB, 0, stream>>>(hbf, wt1, hlin, (int)Nn, HID, HID);
    k_fill<<<4096, TB, 0, stream>>>(agg, 0.0f, nh);
    k_scatter<<<scatGrid, TB, 0, stream>>>(src, dst, dinv, hlin, agg, E);
    k_finalize<<<finGrid, TB, 0, stream>>>(agg, hlin, dinv, b1, nullptr, hbf, Nn, 1);

    // ---- layer 2 (no activation; fp32 out, in-place into agg) ----
    k_gemm<<<gemmGrid, TB, 0, stream>>>(hbf, wt2, hlin, (int)Nn, HID, HID);
    k_fill<<<4096, TB, 0, stream>>>(agg, 0.0f, nh);
    k_scatter<<<scatGrid, TB, 0, stream>>>(src, dst, dinv, hlin, agg, E);
    k_finalize<<<finGrid, TB, 0, stream>>>(agg, hlin, dinv, b2, agg, nullptr, Nn, 0);

    // pool + MLP head
    k_pool<<<B, HID, 0, stream>>>(agg, gbuf);
    k_mlp<<<(B * 128 + TB - 1) / TB, TB, 0, stream>>>(gbuf, fcW0, fcb0, t0, B, 256, 128, 1);
    k_mlp<<<(B * 64 + TB - 1) / TB, TB, 0, stream>>>(t0, fcW1, fcb1, t1, B, 128, 64, 1);
    k_mlp<<<(B * 2 + TB - 1) / TB, TB, 0, stream>>>(t1, fcW2, fcb2, (float*)d_out, B, 64, 2, 0);
}
